// Conv_14096082665597
// MI455X (gfx1250) — matrix operand order repaired and checked by reading, not compiled, not run on hardware
//
#include <hip/hip_runtime.h>

// Conv2d 5x5 pad 2 stride 1: X (32,32,128,128) f32 NCHW, K (64,32,5,5) f32 OIHW
// -> out (32,64,128,128) f32.
// Implicit GEMM on v_wmma_f32_16x16x32_bf16 (K per WMMA = Cin = 32, f32 accum over
// 25 taps). Block = 2 output rows x 128 px x 64 ch, 8 waves, 200 WMMAs/wave.
// Weights repacked to bf16 B-fragment order in d_ws (102400 B), staged into LDS
// per block with global_load_async_to_lds_b128 (ASYNCcnt) overlapped with the
// bf16 X-halo staging.  Inner-loop DS base addresses are pinned in VGPRs so the
// row/tap terms fold into 16-bit DS offset immediates with no rematerialization.

typedef __attribute__((ext_vector_type(16))) __bf16 v16bf;
typedef __attribute__((ext_vector_type(2)))  __bf16 v2bf;
typedef __attribute__((ext_vector_type(8)))  float  v8f;
typedef __attribute__((ext_vector_type(2)))  float  v2f;

union Frag {
    v16bf v;
    uint4 q[2];
};

#define SX_BYTES 50688      // 6 rows * 132 cols * 64B  (also reused as sC)
#define SB_BYTES 102400     // 2 npairs * 50 fragments * 1024B
#define LDS_BYTES (SX_BYTES + SB_BYTES)
#define ROW_BYTES (132 * 64)   // 8448

// ---------------------------------------------------------------------------
// Repack K (64,32,5,5) f32 -> bf16 B fragments, linear layout for verbatim
// async copy ws -> LDS.  Fragment fi = npair*50 + t*2 + nn  (1024B each):
//   bytes [0,512):    lane l*16B = elements i=0..7  -> cin = 16*(l>=16) + i
//   bytes [512,1024): lane l*16B = elements i=8..15 -> cin = 16*(l>=16) + i
//   column N = l&15 -> co = (npair*2+nn)*16 + N
// 51200 bf16 elements -> <<<200, 256>>>.
// ---------------------------------------------------------------------------
__global__ void conv5x5_repack_w(const float* __restrict__ Kw,
                                 __bf16* __restrict__ wsbf) {
    int idx  = blockIdx.x * 256 + threadIdx.x;   // 0..51199
    int fi   = idx >> 9;                         // fragment 0..99
    int rem  = idx & 511;
    int h2   = rem >> 8;                         // element half (i/8)
    int rem2 = rem & 255;
    int l    = rem2 >> 3;                        // lane 0..31
    int j    = rem2 & 7;                         // element within chunk
    int npair = fi / 50;
    int r     = fi % 50;
    int t     = r >> 1;                          // tap 0..24
    int nn    = r & 1;
    int co  = (npair * 2 + nn) * 16 + (l & 15);
    int cin = ((l >> 4) << 4) + h2 * 8 + j;
    int kh = t / 5, kw = t % 5;
    wsbf[idx] = (__bf16)Kw[(co * 32 + cin) * 25 + kh * 5 + kw];
}

// ---------------------------------------------------------------------------
// Main kernel: grid (64 h-blocks, 32 n), block 256 = 8 waves.
// Wave wid: mtiles {2*(wid&3), 2*(wid&3)+1}, ntile pair wid>>2,
// output rows h0, h0+1  -> 8 accumulators, 8 WMMAs per tap.
// ---------------------------------------------------------------------------
__global__ void __launch_bounds__(256)
conv5x5_wmma(const float* __restrict__ X,
             const __bf16* __restrict__ wsbf,
             float* __restrict__ out) {
    extern __shared__ unsigned char smem[];
    unsigned char* sXb = smem;                  // bf16 X tile (chunk-swizzled)
    unsigned char* sB  = smem + SX_BYTES;       // bf16 B fragments
    float*         sC  = (float*)smem;          // epilogue, reuses X region

    const int tid   = threadIdx.x;
    const int lane  = tid & 31;
    const int wid   = tid >> 5;
    const int hi    = (lane >> 4) & 1;
    const int mrow  = lane & 15;
    const int mp    = wid & 3;                  // mtiles 2mp, 2mp+1
    const int npair = wid >> 2;                 // ntiles 2*npair, 2*npair+1
    const int h0    = blockIdx.x * 2;
    const int n     = blockIdx.y;

    // ---- async-stage B fragments: ws -> LDS (102400B, 16B/lane/op) -----------
    {
        unsigned           ldsa = (unsigned)(size_t)sB + tid * 16;
        unsigned long long ga   = (unsigned long long)(size_t)wsbf + tid * 16;
#pragma unroll
        for (int it = 0; it < 25; ++it) {        // 25 * 256 lanes * 16B = 102400B
            asm volatile("global_load_async_to_lds_b128 %0, %1, off"
                         :: "v"(ldsa + it * 4096),
                            "v"(ga + (unsigned long long)(it * 4096))
                         : "memory");
        }
    }

    // ---- stage X halo (6 rows x 132 cols x 32 cin) as bf16, zero-padded ------
    // Swizzled layout: 64B per (row,col); 16B chunk k (cin 8k..8k+7) stored at
    // chunk slot (k + (col>>2)) & 3  -> conflict-free b128 A reads.
    const float* Xn = X + (size_t)n * 32 * 128 * 128;
    for (int p = tid; p < 12672; p += 256) {     // 66 x-pairs * 6 rows * 32 ch
        int xp  = p % 66;
        int rem = p / 66;
        int row = rem % 6;
        int c   = rem / 6;
        int xl  = xp * 2;                        // local col 0..130 (even)
        int y   = h0 - 2 + row;
        int x   = -2 + xl;
        v2f vv = {0.f, 0.f};
        if ((unsigned)y < 128u && x >= 0 && x <= 126) {
            vv = *(const v2f*)(Xn + ((c * 128 + y) * 128 + x));
        }
        v2bf pk = __builtin_convertvector(vv, v2bf);   // v_cvt_pk_bf16_f32
        int k = c >> 3;
        int e = (c & 7) << 1;                    // byte within 16B chunk
        int b0 = (row * 132 + xl)     * 64 + (((k + ( xl      >> 2)) & 3) << 4) + e;
        int b1 = (row * 132 + xl + 1) * 64 + (((k + ((xl + 1) >> 2)) & 3) << 4) + e;
        *(__bf16*)(sXb + b0) = pk[0];
        *(__bf16*)(sXb + b1) = pk[1];
    }

    // ---- hoisted + pinned DS base addresses ----------------------------------
    // A: 2 mtiles x 5 kw x 2 chunks; per-use immediate = (hr+kh)*ROW_BYTES.
    int aoff[2][5][2];
#pragma unroll
    for (int m = 0; m < 2; ++m)
#pragma unroll
        for (int kw = 0; kw < 5; ++kw) {
            int col = (2 * mp + m) * 16 + mrow + kw;
            int q   = (col >> 2) & 3;
            aoff[m][kw][0] = col * 64 + (((2 * hi     + q) & 3) << 4);
            aoff[m][kw][1] = col * 64 + (((2 * hi + 1 + q) & 3) << 4);
            // Pin in VGPRs: forbid rematerialization of the swizzle math in the
            // unrolled tap loop; DS addr becomes reg + offset immediate.
            asm volatile("" : "+v"(aoff[m][kw][0]), "+v"(aoff[m][kw][1]));
        }
    // B: wave base; per-use immediate = (t*2+nn)*1024 (+512) <= 51200 < 65536.
    const unsigned char* bbase = sB + npair * 51200 + lane * 16;

    asm volatile("s_wait_asynccnt 0x0" ::: "memory");
    __syncthreads();

    // ---- 25 taps x (2 B-frags, 4 A-frags, 8 WMMAs) ---------------------------
    v8f acc[2][2][2];                            // [hrow][mtile][ntile]
#pragma unroll
    for (int a = 0; a < 2; ++a)
#pragma unroll
        for (int b = 0; b < 2; ++b)
#pragma unroll
            for (int cc = 0; cc < 2; ++cc)
                acc[a][b][cc] = (v8f){0.f, 0.f, 0.f, 0.f, 0.f, 0.f, 0.f, 0.f};

#pragma unroll
    for (int t = 0; t < 25; ++t) {
        const int kh = t / 5;
        const int kw = t % 5;

        Frag bf[2];
#pragma unroll
        for (int nn = 0; nn < 2; ++nn) {
            bf[nn].q[0] = *(const uint4*)(bbase + (t * 2 + nn) * 1024);
            bf[nn].q[1] = *(const uint4*)(bbase + (t * 2 + nn) * 1024 + 512);
        }

#pragma unroll
        for (int hr = 0; hr < 2; ++hr) {
#pragma unroll
            for (int m = 0; m < 2; ++m) {
                Frag a;   // identical (row = hr+kh) loads across taps CSE away
                a.q[0] = *(const uint4*)(sXb + aoff[m][kw][0] + (hr + kh) * ROW_BYTES);
                a.q[1] = *(const uint4*)(sXb + aoff[m][kw][1] + (hr + kh) * ROW_BYTES);

                acc[hr][m][0] = __builtin_amdgcn_wmma_f32_16x16x32_bf16(
                    false, a.v, false, bf[0].v, (short)0, acc[hr][m][0], false, false);
                acc[hr][m][1] = __builtin_amdgcn_wmma_f32_16x16x32_bf16(
                    false, a.v, false, bf[1].v, (short)0, acc[hr][m][1], false, false);
            }
        }
    }

    // ---- epilogue per output row: C -> LDS (stride 129) -> coalesced float4 --
#pragma unroll
    for (int hr = 0; hr < 2; ++hr) {
        __syncthreads();                          // sX/sB reads (or prev pass) done
        {
            const int co0 = (npair * 2)     * 16 + mrow;
            const int co1 = (npair * 2 + 1) * 16 + mrow;
#pragma unroll
            for (int m = 0; m < 2; ++m) {
                const int wb = (2 * mp + m) * 16 + hi * 8;  // C VGPR r: M = r+hi*8
#pragma unroll
                for (int r = 0; r < 8; ++r) {
                    sC[co0 * 129 + wb + r] = acc[hr][m][0][r];
                    sC[co1 * 129 + wb + r] = acc[hr][m][1][r];
                }
            }
        }
        __syncthreads();

        const int h = h0 + hr;
#pragma unroll
        for (int it = 0; it < 8; ++it) {
            int idx = it * 256 + tid;             // 0..2047 float4s
            int co  = idx >> 5;
            int wq  = idx & 31;
            int b   = co * 129 + wq * 4;
            float4 v;
            v.x = sC[b];
            v.y = sC[b + 1];
            v.z = sC[b + 2];
            v.w = sC[b + 3];
            float* dst = out + (((size_t)n * 64 + co) * 128 + h) * 128 + wq * 4;
            *(float4*)dst = v;
        }
    }
}

extern "C" void kernel_launch(void* const* d_in, const int* in_sizes, int n_in,
                              void* d_out, int out_size, void* d_ws, size_t ws_size,
                              hipStream_t stream) {
    const float* X  = (const float*)d_in[0];       // (32,32,128,128) f32
    const float* Kw = (const float*)d_in[1];       // (64,32,5,5) f32
    float* outp = (float*)d_out;                   // (32,64,128,128) f32
    __bf16* wsbf = (__bf16*)d_ws;                  // 51200 bf16 = 102400 B

    conv5x5_repack_w<<<200, 256, 0, stream>>>(Kw, wsbf);
    dim3 grid(64, 32);
    conv5x5_wmma<<<grid, 256, LDS_BYTES, stream>>>(X, wsbf, outp);
}
